// DynSMHALayer_16853451670043
// MI455X (gfx1250) — matrix operand order repaired and checked by reading, not compiled, not run on hardware
//
#include <hip/hip_runtime.h>

// ---------------------------------------------------------------------------
// DynSMHA layer for MI455X (gfx1250): bf16 WMMA everywhere, f32 accumulate.
// B=2, T=4096, C=1024, D=128, E=8, N=B*T=8192.
// ---------------------------------------------------------------------------

#define B_  2
#define T_  4096
#define C_  1024
#define D_  128
#define E_  8
#define N_  (B_ * T_)

typedef __attribute__((ext_vector_type(16))) __bf16          bf16x16;
typedef __attribute__((ext_vector_type(8)))  float           f32x8;
typedef __attribute__((ext_vector_type(4)))  float           f32x4;
typedef __attribute__((ext_vector_type(8)))  unsigned short  u16x8;

union TileAB {
    bf16x16        v;
    u16x8          h2[2];
    unsigned short u[16];
};

__device__ __forceinline__ unsigned short f32_to_bf16(float f) {
    union { float f; unsigned u; } c; c.f = f;
    unsigned u = c.u;
    u += 0x7FFFu + ((u >> 16) & 1u);   // round-to-nearest-even
    return (unsigned short)(u >> 16);
}

// A fragment: 16x32 bf16. lane holds row M=lane&15; halves h0..7 -> K = 8*hi + 0..7,
// halves h8..15 -> K = 16 + 8*hi + 0..7. `p` already includes (row, k0, hi*8).
__device__ __forceinline__ TileAB load_tile_a(const unsigned short* p) {
    TileAB t;
    t.h2[0] = *(const u16x8*)(p);
    t.h2[1] = *(const u16x8*)(p + 16);
    return t;
}

// B fragment: 32x16 bf16 (KxN). lane holds col N=lane&15; halves h0..7 -> K = 8*hi + 0..7,
// halves h8..15 -> K = 16 + 8*hi + 0..7. `p` includes (row N, k0, hi*8).
__device__ __forceinline__ TileAB load_tile_b(const unsigned short* p) {
    TileAB t;
    t.h2[0] = *(const u16x8*)(p);
    t.h2[1] = *(const u16x8*)(p + 16);
    return t;
}

__device__ __forceinline__ f32x8 wmma_bf16(TileAB a, TileAB b, f32x8 c) {
    return __builtin_amdgcn_wmma_f32_16x16x32_bf16(
        false, a.v, false, b.v, (short)0, c, false, false);
}

// ---------------------------------------------------------------------------
// Kernel 1: elementwise f32 -> bf16
// ---------------------------------------------------------------------------
__global__ void cvt_bf16_kernel(const float* __restrict__ src,
                                unsigned short* __restrict__ dst, int n) {
    for (int i = blockIdx.x * blockDim.x + threadIdx.x; i < n;
         i += gridDim.x * blockDim.x)
        dst[i] = f32_to_bf16(src[i]);
}

// ---------------------------------------------------------------------------
// Kernel 2: per-expert transpose+convert:  src[E,R,S] f32 -> dst[E,S,R] bf16
// ---------------------------------------------------------------------------
__global__ void transpose_cvt_kernel(const float* __restrict__ src,
                                     unsigned short* __restrict__ dst,
                                     int R, int S) {
    int total = E_ * R * S;
    for (int i = blockIdx.x * blockDim.x + threadIdx.x; i < total;
         i += gridDim.x * blockDim.x) {
        int e = i / (R * S);
        int rem = i - e * R * S;
        int r = rem / S;
        int s = rem - r * S;
        dst[(size_t)e * R * S + (size_t)s * R + r] = f32_to_bf16(src[i]);
    }
}

// ---------------------------------------------------------------------------
// Kernel 3: gating. One wave32 per token.
//   logits = normalize(x) @ normalize_cols(sim) - sigmoid(gates)
//   mask = relu>0 (top-2 fallback), weights = mask * softmax(masked gated)
// ---------------------------------------------------------------------------
__global__ void gating_kernel(const float* __restrict__ x,
                              const float* __restrict__ sim,
                              const float* __restrict__ gates,
                              float* __restrict__ m_out,
                              float* __restrict__ w_out) {
    const int lane  = threadIdx.x & 31;
    const int token = blockIdx.x * (blockDim.x >> 5) + (threadIdx.x >> 5);

    float dot[E_], ssq[E_];
    float xsq = 0.0f;
#pragma unroll
    for (int e = 0; e < E_; ++e) { dot[e] = 0.0f; ssq[e] = 0.0f; }

    const float* xr = x + (size_t)token * C_;
    for (int c = lane; c < C_; c += 32) {
        float xv = xr[c];
        xsq += xv * xv;
        const f32x4* sr = (const f32x4*)(sim + (size_t)c * E_);
        f32x4 s0 = sr[0], s1 = sr[1];
        float sv[E_] = { s0.x, s0.y, s0.z, s0.w, s1.x, s1.y, s1.z, s1.w };
#pragma unroll
        for (int e = 0; e < E_; ++e) {
            dot[e] += xv * sv[e];
            ssq[e] += sv[e] * sv[e];
        }
    }
#pragma unroll
    for (int off = 16; off > 0; off >>= 1) {
        xsq += __shfl_xor(xsq, off, 32);
#pragma unroll
        for (int e = 0; e < E_; ++e) {
            dot[e] += __shfl_xor(dot[e], off, 32);
            ssq[e] += __shfl_xor(ssq[e], off, 32);
        }
    }

    const float xden = fmaxf(sqrtf(xsq), 1e-12f);
    float logits[E_], gated[E_], msk[E_];
    float any = 0.0f;
#pragma unroll
    for (int e = 0; e < E_; ++e) {
        float sden = fmaxf(sqrtf(ssq[e]), 1e-12f);
        float sig  = 1.0f / (1.0f + __expf(-gates[e]));
        float lg   = dot[e] / (xden * sden) - sig;
        logits[e]  = lg;
        gated[e]   = fmaxf(lg, 0.0f);
        msk[e]     = (lg > 0.0f) ? 1.0f : 0.0f;
        any       += msk[e];
    }
    if (any == 0.0f) {                       // top-2 fallback on raw logits
        int i1 = 0; float v1 = logits[0];
#pragma unroll
        for (int e = 1; e < E_; ++e)
            if (logits[e] > v1) { v1 = logits[e]; i1 = e; }
        int i2 = -1; float v2 = -3.0e38f;
#pragma unroll
        for (int e = 0; e < E_; ++e)
            if (e != i1 && logits[e] > v2) { v2 = logits[e]; i2 = e; }
#pragma unroll
        for (int e = 0; e < E_; ++e)
            msk[e] = (e == i1 || e == i2) ? 1.0f : 0.0f;
    }
    float ml[E_], mx = -3.0e38f;
#pragma unroll
    for (int e = 0; e < E_; ++e) {
        ml[e] = (msk[e] > 0.0f) ? gated[e] : -1.0e9f;
        mx = fmaxf(mx, ml[e]);
    }
    float sum = 0.0f, pv[E_];
#pragma unroll
    for (int e = 0; e < E_; ++e) { pv[e] = __expf(ml[e] - mx); sum += pv[e]; }

    if (lane < E_) {
        float mv = 0.0f, wv = 0.0f;
#pragma unroll
        for (int e = 0; e < E_; ++e)
            if (e == lane) { mv = msk[e]; wv = msk[e] * pv[e] / sum; }
        m_out[(size_t)token * E_ + lane] = mv;
        w_out[(size_t)token * E_ + lane] = wv;
    }
}

// ---------------------------------------------------------------------------
// Kernel 4: masked grouped QKV GEMM.  One wave per 16x16 tile.
//   which (blockIdx.z): 0 -> Q [t,d], 1 -> K [t,d], 2 -> V transposed [b,d,t]
// ---------------------------------------------------------------------------
__global__ void qkv_kernel(const unsigned short* __restrict__ xbf,
                           const unsigned short* __restrict__ wqT,
                           const unsigned short* __restrict__ wkT,
                           const unsigned short* __restrict__ wvT,
                           const float* __restrict__ mmask,
                           unsigned short* __restrict__ qb,
                           unsigned short* __restrict__ kb,
                           unsigned short* __restrict__ vT) {
    const int lane = threadIdx.x & 31;
    const int mrow = lane & 15;
    const int hi   = lane >> 4;
    const int tM   = blockIdx.x;            // 0..511  (16 tokens each)
    const int tN   = blockIdx.y;            // 0..7    (16 d-cols each)
    const int which = blockIdx.z;           // 0=q 1=k 2=v

    const unsigned short* wT = (which == 0) ? wqT : (which == 1) ? wkT : wvT;

    const f32x8 vzero = {0.f,0.f,0.f,0.f,0.f,0.f,0.f,0.f};
    f32x8 acc[E_];
#pragma unroll
    for (int e = 0; e < E_; ++e) acc[e] = vzero;

    for (int k0 = 0; k0 < C_; k0 += 32) {
        TileAB A = load_tile_a(xbf + (size_t)(tM * 16 + mrow) * C_ + k0 + hi * 8);
#pragma unroll
        for (int e = 0; e < E_; ++e) {
            TileAB Bt = load_tile_b(
                wT + (size_t)(e * D_ + tN * 16 + mrow) * C_ + k0 + hi * 8);
            acc[e] = wmma_bf16(A, Bt, acc[e]);
        }
    }

    const int d = tN * 16 + mrow;
#pragma unroll
    for (int r = 0; r < 8; ++r) {
        int tok = tM * 16 + r + 8 * hi;
        const f32x4* mp = (const f32x4*)(mmask + (size_t)tok * E_);
        f32x4 m0 = mp[0], m1 = mp[1];
        float s = m0.x * acc[0][r] + m0.y * acc[1][r] +
                  m0.z * acc[2][r] + m0.w * acc[3][r] +
                  m1.x * acc[4][r] + m1.y * acc[5][r] +
                  m1.z * acc[6][r] + m1.w * acc[7][r];
        unsigned short h = f32_to_bf16(s);
        if (which == 2) {
            int b  = tok >> 12;            // tok / T_
            int tl = tok & (T_ - 1);
            vT[((size_t)b * D_ + d) * T_ + tl] = h;
        } else if (which == 1) {
            kb[(size_t)tok * D_ + d] = h;
        } else {
            qb[(size_t)tok * D_ + d] = h;
        }
    }
}

// ---------------------------------------------------------------------------
// Kernel 5: causal flash attention. Wave per 16-query tile, 32-key blocks.
//   q,k stored [b*T, D] bf16; v stored transposed [b, D, T] bf16.
// ---------------------------------------------------------------------------
#define LROW 36   // padded LDS row stride (floats): conflict-reduced
__global__ void attn_kernel(const unsigned short* __restrict__ qb,
                            const unsigned short* __restrict__ kb,
                            const unsigned short* __restrict__ vT,
                            unsigned short* __restrict__ ob) {
    __shared__ float plds[2][16 * LROW];

    const int lane = threadIdx.x & 31;
    const int wave = threadIdx.x >> 5;
    const int mrow = lane & 15;
    const int hi   = lane >> 4;
    const int qt   = blockIdx.x * 2 + wave;   // 0..255
    const int b    = blockIdx.y;
    const int q0   = qt * 16;
    const size_t tokq = (size_t)b * T_ + q0;
    const float scale = 0.08838834764831845f; // 1/sqrt(128)

    TileAB Aq[4];
#pragma unroll
    for (int ks = 0; ks < 4; ++ks)
        Aq[ks] = load_tile_a(qb + (tokq + mrow) * D_ + ks * 32 + hi * 8);

    const f32x8 vzero = {0.f,0.f,0.f,0.f,0.f,0.f,0.f,0.f};
    f32x8 O[8];
    float mstat[8], lstat[8];
#pragma unroll
    for (int dt = 0; dt < 8; ++dt) O[dt] = vzero;
#pragma unroll
    for (int r = 0; r < 8; ++r) { mstat[r] = -3.0e38f; lstat[r] = 0.0f; }

    for (int j0 = 0; j0 < q0 + 16; j0 += 32) {
        f32x8 S0 = vzero, S1 = vzero;
#pragma unroll
        for (int ks = 0; ks < 4; ++ks) {
            TileAB Bk0 = load_tile_b(
                kb + ((size_t)b * T_ + j0 + mrow) * D_ + ks * 32 + hi * 8);
            TileAB Bk1 = load_tile_b(
                kb + ((size_t)b * T_ + j0 + 16 + mrow) * D_ + ks * 32 + hi * 8);
            S0 = wmma_bf16(Aq[ks], Bk0, S0);
            S1 = wmma_bf16(Aq[ks], Bk1, S1);
        }
#pragma unroll
        for (int r = 0; r < 8; ++r) {
            int qrow = q0 + r + 8 * hi;
            int kcol = j0 + mrow;
            float s0 = (kcol      <= qrow) ? S0[r] * scale : -3.0e38f;
            float s1 = (kcol + 16 <= qrow) ? S1[r] * scale : -3.0e38f;

            float rmax = fmaxf(s0, s1);
#pragma unroll
            for (int msk = 8; msk > 0; msk >>= 1)
                rmax = fmaxf(rmax, __shfl_xor(rmax, msk, 32));
            float nm   = fmaxf(mstat[r], rmax);
            float corr = __expf(mstat[r] - nm);
            mstat[r]   = nm;
            float p0 = __expf(s0 - nm);
            float p1 = __expf(s1 - nm);
            float rsum = p0 + p1;
#pragma unroll
            for (int msk = 8; msk > 0; msk >>= 1)
                rsum += __shfl_xor(rsum, msk, 32);
            lstat[r] = lstat[r] * corr + rsum;
#pragma unroll
            for (int dt = 0; dt < 8; ++dt) O[dt][r] *= corr;

            plds[wave][(r + 8 * hi) * LROW + mrow]      = p0;
            plds[wave][(r + 8 * hi) * LROW + 16 + mrow] = p1;
        }
        // P (16x32 f32, C-layout in LDS) -> A-layout bf16 fragment
        TileAB Pa;
        const float* l0 = &plds[wave][mrow * LROW + hi * 8];
#pragma unroll
        for (int i = 0; i < 8; ++i) Pa.u[i] = f32_to_bf16(l0[i]);
        const float* l1 = l0 + 16;
#pragma unroll
        for (int i = 0; i < 8; ++i) Pa.u[8 + i] = f32_to_bf16(l1[i]);

#pragma unroll
        for (int dt = 0; dt < 8; ++dt) {
            TileAB Bv = load_tile_b(
                vT + ((size_t)b * D_ + dt * 16 + mrow) * T_ + j0 + hi * 8);
            O[dt] = wmma_bf16(Pa, Bv, O[dt]);
        }
    }

#pragma unroll
    for (int r = 0; r < 8; ++r) {
        float inv = 1.0f / lstat[r];
        size_t tok = tokq + r + 8 * hi;
#pragma unroll
        for (int dt = 0; dt < 8; ++dt)
            ob[tok * D_ + dt * 16 + mrow] = f32_to_bf16(O[dt][r] * inv);
    }
}

// ---------------------------------------------------------------------------
// Kernel 6: weighted output projection. Wave per 16x16 tile of [N, C] output.
// ---------------------------------------------------------------------------
__global__ void outproj_kernel(const unsigned short* __restrict__ ob,
                               const unsigned short* __restrict__ woT,
                               const float* __restrict__ wgt,
                               float* __restrict__ out) {
    const int lane = threadIdx.x & 31;
    const int mrow = lane & 15;
    const int hi   = lane >> 4;
    const int tM   = blockIdx.x;            // 0..511
    const int cN   = blockIdx.y;            // 0..63

    const f32x8 vzero = {0.f,0.f,0.f,0.f,0.f,0.f,0.f,0.f};
    f32x8 acc[E_];
#pragma unroll
    for (int e = 0; e < E_; ++e) acc[e] = vzero;

#pragma unroll
    for (int ks = 0; ks < 4; ++ks) {
        TileAB A = load_tile_a(ob + (size_t)(tM * 16 + mrow) * D_ + ks * 32 + hi * 8);
#pragma unroll
        for (int e = 0; e < E_; ++e) {
            TileAB Bt = load_tile_b(
                woT + ((size_t)e * C_ + cN * 16 + mrow) * D_ + ks * 32 + hi * 8);
            acc[e] = wmma_bf16(A, Bt, acc[e]);
        }
    }

#pragma unroll
    for (int r = 0; r < 8; ++r) {
        int tok = tM * 16 + r + 8 * hi;
        const f32x4* wp = (const f32x4*)(wgt + (size_t)tok * E_);
        f32x4 w0 = wp[0], w1 = wp[1];
        float s = w0.x * acc[0][r] + w0.y * acc[1][r] +
                  w0.z * acc[2][r] + w0.w * acc[3][r] +
                  w1.x * acc[4][r] + w1.y * acc[5][r] +
                  w1.z * acc[6][r] + w1.w * acc[7][r];
        out[(size_t)tok * C_ + cN * 16 + mrow] = s;
    }
}

// ---------------------------------------------------------------------------
// Launch
// ---------------------------------------------------------------------------
extern "C" void kernel_launch(void* const* d_in, const int* in_sizes, int n_in,
                              void* d_out, int out_size, void* d_ws, size_t ws_size,
                              hipStream_t stream) {
    (void)in_sizes; (void)n_in; (void)out_size; (void)ws_size;
    const float* x   = (const float*)d_in[0];   // [B,T,C]
    const float* sim = (const float*)d_in[1];   // [C,E]
    const float* gts = (const float*)d_in[2];   // [E]
    const float* qp  = (const float*)d_in[3];   // [E,C,D]
    const float* kp  = (const float*)d_in[4];
    const float* vp  = (const float*)d_in[5];
    const float* op  = (const float*)d_in[6];   // [E,D,C]
    float* out = (float*)d_out;                 // [B,T,C] f32

    char* ws = (char*)d_ws;
    size_t off = 0;
    auto take = [&](size_t bytes) -> char* {
        char* p = ws + off;
        off = (off + bytes + 255) & ~(size_t)255;
        return p;
    };
    float*          m_ws = (float*)take((size_t)N_ * E_ * 4);
    float*          w_ws = (float*)take((size_t)N_ * E_ * 4);
    unsigned short* xbf  = (unsigned short*)take((size_t)N_ * C_ * 2);
    unsigned short* qb   = (unsigned short*)take((size_t)N_ * D_ * 2);
    unsigned short* kb   = (unsigned short*)take((size_t)N_ * D_ * 2);
    unsigned short* vT   = (unsigned short*)take((size_t)N_ * D_ * 2);
    unsigned short* ob   = (unsigned short*)take((size_t)N_ * D_ * 2);
    unsigned short* wqT  = (unsigned short*)take((size_t)E_ * C_ * D_ * 2);
    unsigned short* wkT  = (unsigned short*)take((size_t)E_ * C_ * D_ * 2);
    unsigned short* wvT  = (unsigned short*)take((size_t)E_ * C_ * D_ * 2);
    unsigned short* woT  = (unsigned short*)take((size_t)E_ * C_ * D_ * 2);

    cvt_bf16_kernel<<<4096, 256, 0, stream>>>(x, xbf, N_ * C_);
    transpose_cvt_kernel<<<2048, 256, 0, stream>>>(qp, wqT, C_, D_); // -> [E,D,C]
    transpose_cvt_kernel<<<2048, 256, 0, stream>>>(kp, wkT, C_, D_);
    transpose_cvt_kernel<<<2048, 256, 0, stream>>>(vp, wvT, C_, D_);
    transpose_cvt_kernel<<<2048, 256, 0, stream>>>(op, woT, D_, C_); // -> [E,C,D]

    gating_kernel<<<N_ / 4, 128, 0, stream>>>(x, sim, gts, m_ws, w_ws);

    qkv_kernel<<<dim3(N_ / 16, D_ / 16, 3), 32, 0, stream>>>(
        xbf, wqT, wkT, wvT, m_ws, qb, kb, vT);

    attn_kernel<<<dim3(T_ / 32, B_), 64, 0, stream>>>(qb, kb, vT, ob);

    outproj_kernel<<<dim3(N_ / 16, C_ / 16), 32, 0, stream>>>(ob, woT, w_ws, out);
}
